// NeuralFingerprint_61040075211144
// MI455X (gfx1250) — matrix operand order repaired and checked by reading, not compiled, not run on hardware
//
#include <hip/hip_runtime.h>
#include <hip/hip_bf16.h>

typedef __attribute__((ext_vector_type(16))) __bf16 v16bf;
typedef __attribute__((ext_vector_type(8)))  float  v8f;
typedef __attribute__((ext_vector_type(4)))  int    v4i;

typedef __attribute__((address_space(1))) v4i glob_v4i;
typedef __attribute__((address_space(3))) v4i lds_v4i;

#define HID 64
#define MAXK 128
#define WAVES 4                 // waves per block in GEMM
#define ASTRIDE (MAXK + 4)      // padded LDS row stride (floats)

// ---------------------------------------------------------------------------
// Degree / normalization kernels
// ---------------------------------------------------------------------------
__global__ void deg_init_kernel(float* __restrict__ deg, int N) {
    int i = blockIdx.x * blockDim.x + threadIdx.x;
    if (i < N) deg[i] = 1.0f;  // self loop
}

__global__ void deg_count_kernel(const long long* __restrict__ rows,
                                 float* __restrict__ deg, int E) {
    int e = blockIdx.x * blockDim.x + threadIdx.x;
    if (e < E) atomicAdd(&deg[rows[e]], 1.0f);
}

__global__ void dis_kernel(const float* __restrict__ deg,
                           float* __restrict__ dis, int N) {
    int i = blockIdx.x * blockDim.x + threadIdx.x;
    if (i < N) dis[i] = rsqrtf(deg[i]);
}

// ---------------------------------------------------------------------------
// Pre-pack a [K,64] f32 weight matrix into per-lane bf16 WMMA B-fragments.
// Fragment (chunk,tile): lane holds 16 bf16 (one v16bf) stored contiguously,
// so the GEMM reads each B fragment as a single coalesced 32B vector load.
// One wave per fragment; frag = chunk*4 + tile.
// ---------------------------------------------------------------------------
__global__ void pack_w_frags_kernel(const float* __restrict__ W,
                                    __bf16* __restrict__ out, int K) {
    int frag  = blockIdx.x;          // (K/32)*4 frags
    int lane  = threadIdx.x;         // 32 threads
    int chunk = frag >> 2;
    int tile  = frag & 3;
    int n     = tile * 16 + (lane & 15);
    int khalf = (lane >> 4) * 16;    // hi lanes hold K=16..31 of the chunk
    int k0    = chunk * 32;
    __bf16* dst = out + ((size_t)frag * 32 + lane) * 16;
#pragma unroll
    for (int v = 0; v < 8; ++v) {
        int kb = khalf + 2 * v;
        dst[2 * v]     = (__bf16)W[(size_t)(k0 + kb) * HID + n];
        dst[2 * v + 1] = (__bf16)W[(size_t)(k0 + kb + 1) * HID + n];
    }
}

// ---------------------------------------------------------------------------
// WMMA GEMM:  out[M,64] = (relu?)(A[M,K] @ W[K,64] + bias)
// One wave computes a 16x64 tile: A tile staged f32 in LDS (async copy when
// available), fragments converted to bf16 from LDS; B fragments are
// pre-packed bf16 loaded as one 32B vector per lane.
// ---------------------------------------------------------------------------
__global__ void gemm_bf16_wmma_n64(const float* __restrict__ A,
                                   const __bf16* __restrict__ Wp,
                                   const float* __restrict__ bias,
                                   float* __restrict__ out,
                                   int M, int K, int do_relu) {
    __shared__ float atile[WAVES * 16 * ASTRIDE];  // ~33.8 KB

    const int lane = threadIdx.x & 31;
    const int wave = threadIdx.x >> 5;
    const int tile = blockIdx.x * (blockDim.x >> 5) + wave;  // wave-uniform
    if (tile * 16 >= M) return;  // whole-wave exit -> EXEC all ones for WMMA

    const int row0 = tile * 16;
    float* lbase = atile + wave * 16 * ASTRIDE;

    // ---- stage 16xK f32 tile into LDS, coalesced 16B per lane ----
    const int segsPerRow = K >> 2;                  // 16B segments per row
    const int lgSeg = (K == 128) ? 5 : 4;           // log2(segsPerRow)
    const int nChunks = K >> 3;                     // (16*K/4)/32 instructions
    for (int i = 0; i < nChunks; ++i) {
        int seg = i * 32 + lane;
        int row = seg >> lgSeg;
        int off = (seg & (segsPerRow - 1)) << 2;    // float offset in row
        const float* gp = A + (size_t)(row0 + row) * K + off;
        float* lp = lbase + row * ASTRIDE + off;
#if __has_builtin(__builtin_amdgcn_global_load_async_to_lds_b128)
        __builtin_amdgcn_global_load_async_to_lds_b128(
            (glob_v4i*)gp, (lds_v4i*)lp, 0, 0);
#else
        *(float4*)lp = *(const float4*)gp;
#endif
    }
#if __has_builtin(__builtin_amdgcn_global_load_async_to_lds_b128)
#if __has_builtin(__builtin_amdgcn_s_wait_asynccnt)
    __builtin_amdgcn_s_wait_asynccnt(0);
#else
    asm volatile("s_wait_asynccnt 0x0" ::: "memory");
#endif
#endif

    const int m      = lane & 15;        // A fragment row
    const int khalfA = (lane >> 4) * 16; // A: hi lanes hold K=16..31 of the chunk

    v8f acc[4] = {};  // 4 N-tiles of f32 16x16 accumulators

    for (int k0 = 0; k0 < K; k0 += 32) {
        // --- A fragment: 16x32 bf16 from LDS (contiguous k = 16*(lane/16) + s, same as packed B) ---
        v16bf a;
        const float* arow = lbase + m * ASTRIDE + k0;
#pragma unroll
        for (int v = 0; v < 8; ++v) {
            int kb = khalfA + 2 * v;
            float2 f2 = *(const float2*)(arow + kb);   // ds_load_b64
            a[2 * v]     = (__bf16)f2.x;
            a[2 * v + 1] = (__bf16)f2.y;
        }
        const int chunk = k0 >> 5;
#pragma unroll
        for (int t = 0; t < 4; ++t) {
            // --- B fragment: pre-packed, one 32B vector per lane ---
            v16bf b = *(const v16bf*)(
                Wp + ((size_t)(chunk * 4 + t) * 32 + lane) * 16);
            acc[t] = __builtin_amdgcn_wmma_f32_16x16x32_bf16(
                false, a, false, b, (short)0, acc[t], false, false);
        }
    }

    // --- store: f32 C/D layout (lane<16: M=v, N=lane; lane>=16: M=8+v) ---
    const int mbase = (lane >> 4) << 3;
#pragma unroll
    for (int t = 0; t < 4; ++t) {
        int nn = t * 16 + (lane & 15);
        float bv = bias ? bias[nn] : 0.0f;
#pragma unroll
        for (int v = 0; v < 8; ++v) {
            float val = acc[t][v] + bv;
            if (do_relu) val = fmaxf(val, 0.0f);
            out[(size_t)(row0 + mbase + v) * HID + nn] = val;
        }
    }
}

// ---------------------------------------------------------------------------
// agg[n,:] = t[n,:] * dis[n]^2   (self-loop term; also initializes agg)
// ---------------------------------------------------------------------------
__global__ void selfloop_init_kernel(const float* __restrict__ t,
                                     const float* __restrict__ deg,
                                     float* __restrict__ agg, int total) {
    int i = blockIdx.x * blockDim.x + threadIdx.x;
    if (i < total) {
        int n = i >> 6;               // /HID
        agg[i] = t[i] / deg[n];       // dis^2 == 1/deg
    }
}

// ---------------------------------------------------------------------------
// Edge scatter: one wave per edge, each lane handles 2 features.
// agg[c,:] += dis[r]*dis[c] * t[r,:]
// ---------------------------------------------------------------------------
__global__ void scatter_edges_kernel(const long long* __restrict__ rows,
                                     const long long* __restrict__ cols,
                                     const float* __restrict__ dis,
                                     const float* __restrict__ t,
                                     float* __restrict__ agg, int E) {
    int gid  = blockIdx.x * blockDim.x + threadIdx.x;
    int edge = gid >> 5;
    int lane = threadIdx.x & 31;
    if (edge >= E) return;
    long long r = rows[edge];
    long long c = cols[edge];
    float nrm = dis[r] * dis[c];
    const float2 v = *(const float2*)(t + (size_t)r * HID + lane * 2);
    atomicAdd(&agg[(size_t)c * HID + lane * 2],     nrm * v.x);
    atomicAdd(&agg[(size_t)c * HID + lane * 2 + 1], nrm * v.y);
}

// ---------------------------------------------------------------------------
// Final projection: out[n] = h2[n,:] . W_o + b_o
// ---------------------------------------------------------------------------
__global__ void out_proj_kernel(const float* __restrict__ h2,
                                const float* __restrict__ Wo,
                                const float* __restrict__ bo,
                                float* __restrict__ out, int N) {
    int i = blockIdx.x * blockDim.x + threadIdx.x;
    if (i >= N) return;
    const float* row = h2 + (size_t)i * HID;
    float s = bo[0];
#pragma unroll
    for (int k = 0; k < HID; ++k) s += row[k] * Wo[k];
    out[i] = s;
}

// ---------------------------------------------------------------------------
extern "C" void kernel_launch(void* const* d_in, const int* in_sizes, int n_in,
                              void* d_out, int out_size, void* d_ws, size_t ws_size,
                              hipStream_t stream) {
    const float*     x    = (const float*)d_in[0];
    const long long* ei   = (const long long*)d_in[1];   // [2, E] int64
    const float*     W_i  = (const float*)d_in[2];
    const float*     b_i  = (const float*)d_in[3];
    const float*     W_c  = (const float*)d_in[4];
    const float*     W_h  = (const float*)d_in[5];
    const float*     b_h  = (const float*)d_in[6];
    const float*     W_o  = (const float*)d_in[7];
    const float*     b_o  = (const float*)d_in[8];
    float* out = (float*)d_out;

    const int N = in_sizes[0] / 128;   // 50000
    const int E = in_sizes[1] / 2;     // 800000
    const int K_IN = in_sizes[0] / N;  // 128

    const long long* rows = ei;
    const long long* cols = ei + E;

    // workspace carve-up (256B aligned)
    char* ws = (char*)d_ws;
    size_t off = 0;
    auto carve = [&](size_t bytes) {
        char* p = ws + off;
        off += (bytes + 255) & ~(size_t)255;
        return p;
    };
    float*  deg  = (float*)carve((size_t)N * sizeof(float));
    float*  dis  = (float*)carve((size_t)N * sizeof(float));
    float*  h1   = (float*)carve((size_t)N * HID * sizeof(float));
    float*  t    = (float*)carve((size_t)N * HID * sizeof(float));
    float*  agg  = (float*)carve((size_t)N * HID * sizeof(float));
    __bf16* Wp_i = (__bf16*)carve((size_t)(K_IN / 32) * 4 * 32 * 16 * sizeof(__bf16));
    __bf16* Wp_c = (__bf16*)carve((size_t)(HID  / 32) * 4 * 32 * 16 * sizeof(__bf16));
    __bf16* Wp_h = (__bf16*)carve((size_t)(HID  / 32) * 4 * 32 * 16 * sizeof(__bf16));
    float*  h2   = h1;  // h1 dead after t is computed

    const int TPB = 256;

    // 0) pre-pack weights into bf16 WMMA B-fragment order (tiny, once per call)
    pack_w_frags_kernel<<<(K_IN / 32) * 4, 32, 0, stream>>>(W_i, Wp_i, K_IN);
    pack_w_frags_kernel<<<(HID  / 32) * 4, 32, 0, stream>>>(W_c, Wp_c, HID);
    pack_w_frags_kernel<<<(HID  / 32) * 4, 32, 0, stream>>>(W_h, Wp_h, HID);

    // 1) degrees (self loops included)
    deg_init_kernel<<<(N + TPB - 1) / TPB, TPB, 0, stream>>>(deg, N);
    deg_count_kernel<<<(E + TPB - 1) / TPB, TPB, 0, stream>>>(rows, deg, E);
    dis_kernel<<<(N + TPB - 1) / TPB, TPB, 0, stream>>>(deg, dis, N);

    // 2) h1 = relu(x @ W_i + b_i)   [N,128]x[128,64]  (WMMA bf16)
    {
        int tiles  = (N + 15) / 16;
        int blocks = (tiles + WAVES - 1) / WAVES;
        gemm_bf16_wmma_n64<<<blocks, WAVES * 32, 0, stream>>>(
            x, Wp_i, b_i, h1, N, K_IN, 1);
    }

    // 3) t = h1 @ W_c   [N,64]x[64,64]  (WMMA bf16)
    {
        int tiles  = (N + 15) / 16;
        int blocks = (tiles + WAVES - 1) / WAVES;
        gemm_bf16_wmma_n64<<<blocks, WAVES * 32, 0, stream>>>(
            h1, Wp_c, nullptr, t, N, HID, 0);
    }

    // 4) agg = self-loop term, then scatter edges
    selfloop_init_kernel<<<((N * HID) + TPB - 1) / TPB, TPB, 0, stream>>>(
        t, deg, agg, N * HID);
    {
        long long threads = (long long)E * 32;
        int blocks = (int)((threads + TPB - 1) / TPB);
        scatter_edges_kernel<<<blocks, TPB, 0, stream>>>(rows, cols, dis, t, agg, E);
    }

    // 5) h2 = relu(agg @ W_h + b_h)  (WMMA bf16)
    {
        int tiles  = (N + 15) / 16;
        int blocks = (tiles + WAVES - 1) / WAVES;
        gemm_bf16_wmma_n64<<<blocks, WAVES * 32, 0, stream>>>(
            agg, Wp_h, b_h, h2, N, HID, 1);
    }

    // 6) out = h2 @ W_o + b_o
    out_proj_kernel<<<(N + TPB - 1) / TPB, TPB, 0, stream>>>(h2, W_o, b_o, out, N);
}
